// RNN_V2_22763326669285
// MI455X (gfx1250) — matrix operand order repaired and checked by reading, not compiled, not run on hardware
//
#include <hip/hip_runtime.h>
#include <cstddef>

// ---------------------------------------------------------------------------
// Types for CDNA5 WMMA (wave32, 16x16x32 bf16 -> f32)
// ---------------------------------------------------------------------------
typedef __attribute__((ext_vector_type(16))) __bf16 v16bf;
typedef __attribute__((ext_vector_type(8)))  __bf16 v8bf;
typedef __attribute__((ext_vector_type(8)))  float  v8f;

// ---------------------------------------------------------------------------
// Problem dims
// ---------------------------------------------------------------------------
#define S_LEN 256
#define BATCH 512
#define FDIM  64
#define HDIM  256
#define G3H   768   // 3*H gate width

// ---------------------------------------------------------------------------
// bf16 weight pool layout (element offsets). Torch layout (out x in) row-major.
// ---------------------------------------------------------------------------
constexpr size_t SZ_IHF  = (size_t)G3H * FDIM;   // 49152
constexpr size_t SZ_HH   = (size_t)G3H * HDIM;   // 196608
constexpr size_t SZ_FC   = (size_t)HDIM * HDIM;  // 65536
constexpr size_t SZ_FC4A = (size_t)FDIM * HDIM;  // 16384
constexpr size_t SZ_FC4B = (size_t)FDIM * FDIM;  // 4096
constexpr size_t SZ_W4   = (size_t)16 * HDIM;    // 4096 (padded head W4)

constexpr size_t OW_G1IH = 0;
constexpr size_t OW_G1HH = OW_G1IH + SZ_IHF;
constexpr size_t OW_G2IH = OW_G1HH + SZ_HH;
constexpr size_t OW_G2HH = OW_G2IH + SZ_HH;
constexpr size_t OW_G3IH = OW_G2HH + SZ_HH;
constexpr size_t OW_G3HH = OW_G3IH + SZ_HH;
constexpr size_t OW_G4IH = OW_G3HH + SZ_HH;
constexpr size_t OW_G4HH = OW_G4IH + SZ_IHF;
constexpr size_t OW_FC1A = OW_G4HH + SZ_HH;
constexpr size_t OW_FC1B = OW_FC1A + SZ_FC;
constexpr size_t OW_FC2A = OW_FC1B + SZ_FC;
constexpr size_t OW_FC2B = OW_FC2A + SZ_FC;
constexpr size_t OW_FC3A = OW_FC2B + SZ_FC;
constexpr size_t OW_FC3B = OW_FC3A + SZ_FC;
constexpr size_t OW_FC4A = OW_FC3B + SZ_FC;
constexpr size_t OW_FC4B = OW_FC4A + SZ_FC4A;
constexpr size_t OW_C1W1 = OW_FC4B + SZ_FC4B;
constexpr size_t OW_C1W2 = OW_C1W1 + SZ_FC;
constexpr size_t OW_C1W3 = OW_C1W2 + SZ_FC;
constexpr size_t OW_C1W4 = OW_C1W3 + SZ_FC;
constexpr size_t OW_C2W1 = OW_C1W4 + SZ_W4;
constexpr size_t OW_C2W2 = OW_C2W1 + SZ_FC;
constexpr size_t OW_C2W3 = OW_C2W2 + SZ_FC;
constexpr size_t OW_C2W4 = OW_C2W3 + SZ_FC;
constexpr size_t POOL_ELEMS = OW_C2W4 + SZ_W4;

// ---------------------------------------------------------------------------
// Core tile GEMM: C(16xN, f32 in LDS) = A(16xK bf16, LDS) @ W(NxK bf16, L2)^T + bias
// K is compile-time: all A-fragments are loaded to registers once and reused
// across the wave's N-tiles; per tile the B-load + WMMA chain is fully
// unrolled so loads clause ahead of the WMMA chain.
// A-fragment: ISA 16-bit A layout -> two b128 LDS loads per lane per K-block.
// B-fragment: ISA 16-bit B layout -> two b128 loads of W row per lane per K-block.
// ---------------------------------------------------------------------------
template <int K>
__device__ __forceinline__ void tile_gemm(const __bf16* __restrict__ A,
                                          const __bf16* __restrict__ W,
                                          const float* __restrict__ bias, int nvalid,
                                          float* __restrict__ C, int N,
                                          int wave, int nwaves, int lane) {
  constexpr int KB = K / 32;
  const int mcol = lane & 15;   // A row / B column / D column
  const int kh   = lane >> 4;   // which K-half of the fragment this lane holds
  const __bf16* arow = A + mcol * K + kh * 8;

  v16bf afrag[KB];
#pragma unroll
  for (int kb = 0; kb < KB; ++kb) {
    v8bf alo = *(const v8bf*)(arow + kb * 32);
    v8bf ahi = *(const v8bf*)(arow + kb * 32 + 16);
    afrag[kb] = __builtin_shufflevector(alo, ahi, 0, 1, 2, 3, 4, 5, 6, 7,
                                        8, 9, 10, 11, 12, 13, 14, 15);
  }

  for (int n0 = wave * 16; n0 < N; n0 += nwaves * 16) {
    const __bf16* wrow = W + (size_t)(n0 + mcol) * K + kh * 8;
    // Pull the next tile's weight lines toward the WGP while this tile's
    // WMMA chain runs (lowers to global_prefetch_b8; no counter impact).
    if (n0 + nwaves * 16 < N)
      __builtin_prefetch(wrow + (size_t)nwaves * 16 * K, 0, 1);
    v8f acc = {0.f, 0.f, 0.f, 0.f, 0.f, 0.f, 0.f, 0.f};
#pragma unroll
    for (int kb = 0; kb < KB; ++kb) {
      v8bf blo = *(const v8bf*)(wrow + kb * 32);
      v8bf bhi = *(const v8bf*)(wrow + kb * 32 + 16);
      v16bf b = __builtin_shufflevector(blo, bhi, 0, 1, 2, 3, 4, 5, 6, 7,
                                        8, 9, 10, 11, 12, 13, 14, 15);
      acc = __builtin_amdgcn_wmma_f32_16x16x32_bf16(false, afrag[kb], false, b,
                                                    (short)0, acc, false, false);
    }
    const int n = n0 + mcol;
    const float bv = (n < nvalid) ? bias[n] : 0.f;  // per-lane cndmask, no branch
    const int m0 = kh * 8;
#pragma unroll
    for (int v = 0; v < 8; ++v) C[(m0 + v) * N + n] = acc[v] + bv;
  }
}

__device__ __forceinline__ void relu_to_bf(const float* __restrict__ s,
                                           __bf16* __restrict__ d, int n,
                                           int tid, int nthr) {
  for (int i = tid; i < n; i += nthr) d[i] = (__bf16)fmaxf(s[i], 0.f);
}

// Row LayerNorm over 256 cols (one wave per row group), optional ReLU, bf16 out.
__device__ __forceinline__ void ln_to_bf(const float* __restrict__ in,
                                         const float* __restrict__ g,
                                         const float* __restrict__ b,
                                         bool relu, __bf16* __restrict__ out,
                                         int wave, int nwaves, int lane) {
  for (int row = wave; row < 16; row += nwaves) {
    const float* rp = in + row * HDIM;
    float s = 0.f, s2 = 0.f;
    for (int j = lane; j < HDIM; j += 32) { float v = rp[j]; s += v; s2 += v * v; }
#pragma unroll
    for (int o = 16; o > 0; o >>= 1) {
      s += __shfl_xor(s, o, 32);
      s2 += __shfl_xor(s2, o, 32);
    }
    const float mu  = s * (1.f / HDIM);
    const float var = s2 * (1.f / HDIM) - mu * mu;
    const float inv = rsqrtf(var + 1e-5f);
    for (int j = lane; j < HDIM; j += 32) {
      float v = (rp[j] - mu) * inv * g[j] + b[j];
      if (relu) v = fmaxf(v, 0.f);
      out[row * HDIM + j] = (__bf16)v;
    }
  }
}

// GRU elementwise combine: gates (r,z,n) from gx/gh (16x768), h bf16 in/out.
__device__ __forceinline__ void gru_combine(const float* __restrict__ gx,
                                            const float* __restrict__ gh,
                                            __bf16* __restrict__ h, int tid,
                                            __bf16* __restrict__ outb,
                                            int t, int b0) {
  for (int i = tid; i < 16 * HDIM; i += 512) {
    const int m = i >> 8, j = i & 255;
    const float* gxm = gx + m * G3H;
    const float* ghm = gh + m * G3H;
    const float r = 1.f / (1.f + __expf(-(gxm[j] + ghm[j])));
    const float z = 1.f / (1.f + __expf(-(gxm[HDIM + j] + ghm[HDIM + j])));
    const float n = tanhf(gxm[2 * HDIM + j] + r * ghm[2 * HDIM + j]);
    const float hv = (1.f - z) * n + z * (float)h[i];
    const __bf16 hq = (__bf16)hv;
    h[i] = hq;
    if (outb) outb[((size_t)t * BATCH + b0 + m) * HDIM + j] = hq;
  }
}

// ---------------------------------------------------------------------------
// Prep kernels
// ---------------------------------------------------------------------------
__global__ void cvt_bf16_kernel(const float* __restrict__ s, __bf16* __restrict__ d, int n) {
  int i = blockIdx.x * blockDim.x + threadIdx.x;
  if (i < n) d[i] = (__bf16)s[i];
}

__global__ void cvt_pad_kernel(const float* __restrict__ s, __bf16* __restrict__ d,
                               int rows, int cols) {
  int i = blockIdx.x * blockDim.x + threadIdx.x;
  if (i < 16 * cols) d[i] = (i / cols < rows) ? (__bf16)s[i] : (__bf16)0.f;
}

__global__ void bn_stats_kernel(const float* __restrict__ X,
                                const float* __restrict__ g,
                                const float* __restrict__ b,
                                float* __restrict__ scale,
                                float* __restrict__ shift) {
  const int f = blockIdx.x, tid = threadIdx.x;
  float s = 0.f, s2 = 0.f;
  for (int i = tid; i < S_LEN * BATCH; i += 256) {
    float v = X[(size_t)i * FDIM + f];
    s += v; s2 += v * v;
  }
  __shared__ float sh[256], sh2[256];
  sh[tid] = s; sh2[tid] = s2;
  __syncthreads();
  for (int o = 128; o > 0; o >>= 1) {
    if (tid < o) { sh[tid] += sh[tid + o]; sh2[tid] += sh2[tid + o]; }
    __syncthreads();
  }
  if (tid == 0) {
    const float inv_n = 1.f / (float)(S_LEN * BATCH);
    const float mu = sh[0] * inv_n;
    const float var = sh2[0] * inv_n - mu * mu;
    const float sc = g[f] * rsqrtf(var + 1e-5f);
    scale[f] = sc;
    shift[f] = b[f] - mu * sc;
  }
}

// ---------------------------------------------------------------------------
// Persistent scan kernel: 32 WGs x 512 threads; WG = 16 batch rows, all 256 steps.
// ---------------------------------------------------------------------------
struct ScanArgs {
  const float* X;
  const float* bnscale;
  const float* bnshift;
  const __bf16* wpool;
  const float* bih[4];
  const float* bhh[4];
  const float* fcb1[4];  // fc1..fc4 first bias
  const float* fcb2[4];  // fc1..fc4 second bias
  const float* lng[3];
  const float* lnb[3];
  __bf16* outb;          // (S*B, H) bf16 scan outputs
};

__global__ void __launch_bounds__(512) scan_kernel(ScanArgs a) {
  __shared__ __attribute__((aligned(16))) float  gx[16 * G3H];     // 48 KB
  __shared__ __attribute__((aligned(16))) float  gh[16 * G3H];     // 48 KB
  __shared__ __attribute__((aligned(16))) __bf16 hb[4][16 * HDIM]; // 32 KB
  __shared__ __attribute__((aligned(16))) __bf16 xb[16 * FDIM];    // 2 KB
  __shared__ __attribute__((aligned(16))) float  xf[16 * FDIM];    // 4 KB
  __shared__ __attribute__((aligned(16))) __bf16 ab[16 * HDIM];    // 8 KB
  __shared__ __attribute__((aligned(16))) __bf16 ab64[16 * FDIM];  // 2 KB

  const int tid = threadIdx.x, lane = tid & 31, wave = tid >> 5;
  const int b0 = blockIdx.x * 16;
  const __bf16* wp = a.wpool;

  for (int i = tid; i < 4 * 16 * HDIM; i += 512) (&hb[0][0])[i] = (__bf16)0.f;
  __syncthreads();

  for (int t = 0; t < S_LEN; ++t) {
    // ---- x_t with BatchNorm affine ----
    for (int i = tid; i < 16 * FDIM; i += 512) {
      const int m = i >> 6, f = i & 63;
      float v = a.X[((size_t)t * BATCH + b0 + m) * FDIM + f];
      v = v * a.bnscale[f] + a.bnshift[f];
      xf[i] = v;
      xb[i] = (__bf16)v;
    }
    __syncthreads();

    // ---- GRU1 ----
    tile_gemm<FDIM>(xb, wp + OW_G1IH, a.bih[0], G3H, gx, G3H, wave, 16, lane);
    tile_gemm<HDIM>(hb[0], wp + OW_G1HH, a.bhh[0], G3H, gh, G3H, wave, 16, lane);
    __syncthreads();
    gru_combine(gx, gh, hb[0], tid, nullptr, 0, 0);
    __syncthreads();
    // ---- FC1 -> LN1 -> ReLU ----
    tile_gemm<HDIM>(hb[0], wp + OW_FC1A, a.fcb1[0], HDIM, gx, HDIM, wave, 16, lane);
    __syncthreads();
    relu_to_bf(gx, ab, 16 * HDIM, tid, 512);
    __syncthreads();
    tile_gemm<HDIM>(ab, wp + OW_FC1B, a.fcb2[0], HDIM, gh, HDIM, wave, 16, lane);
    __syncthreads();
    ln_to_bf(gh, a.lng[0], a.lnb[0], true, ab, wave, 16, lane);
    __syncthreads();

    // ---- GRU2 ----
    tile_gemm<HDIM>(ab, wp + OW_G2IH, a.bih[1], G3H, gx, G3H, wave, 16, lane);
    tile_gemm<HDIM>(hb[1], wp + OW_G2HH, a.bhh[1], G3H, gh, G3H, wave, 16, lane);
    __syncthreads();
    gru_combine(gx, gh, hb[1], tid, nullptr, 0, 0);
    __syncthreads();
    // ---- FC2 -> LN2 -> ReLU ----
    tile_gemm<HDIM>(hb[1], wp + OW_FC2A, a.fcb1[1], HDIM, gx, HDIM, wave, 16, lane);
    __syncthreads();
    relu_to_bf(gx, ab, 16 * HDIM, tid, 512);
    __syncthreads();
    tile_gemm<HDIM>(ab, wp + OW_FC2B, a.fcb2[1], HDIM, gh, HDIM, wave, 16, lane);
    __syncthreads();
    ln_to_bf(gh, a.lng[1], a.lnb[1], true, ab, wave, 16, lane);
    __syncthreads();

    // ---- GRU3 ----
    tile_gemm<HDIM>(ab, wp + OW_G3IH, a.bih[2], G3H, gx, G3H, wave, 16, lane);
    tile_gemm<HDIM>(hb[2], wp + OW_G3HH, a.bhh[2], G3H, gh, G3H, wave, 16, lane);
    __syncthreads();
    gru_combine(gx, gh, hb[2], tid, nullptr, 0, 0);
    __syncthreads();
    // ---- FC3 -> LN3 (no ReLU) ----
    tile_gemm<HDIM>(hb[2], wp + OW_FC3A, a.fcb1[2], HDIM, gx, HDIM, wave, 16, lane);
    __syncthreads();
    relu_to_bf(gx, ab, 16 * HDIM, tid, 512);
    __syncthreads();
    tile_gemm<HDIM>(ab, wp + OW_FC3B, a.fcb2[2], HDIM, gh, HDIM, wave, 16, lane);
    __syncthreads();
    ln_to_bf(gh, a.lng[2], a.lnb[2], false, ab, wave, 16, lane);
    __syncthreads();

    // ---- FC4: H -> F, ReLU inside; residual + x_t ----
    tile_gemm<HDIM>(ab, wp + OW_FC4A, a.fcb1[3], FDIM, gx, FDIM, wave, 16, lane);
    __syncthreads();
    relu_to_bf(gx, ab64, 16 * FDIM, tid, 512);
    __syncthreads();
    tile_gemm<FDIM>(ab64, wp + OW_FC4B, a.fcb2[3], FDIM, gh, FDIM, wave, 16, lane);
    __syncthreads();
    for (int i = tid; i < 16 * FDIM; i += 512) ab64[i] = (__bf16)(gh[i] + xf[i]);
    __syncthreads();

    // ---- GRU4 (writes h4 to global) ----
    tile_gemm<FDIM>(ab64, wp + OW_G4IH, a.bih[3], G3H, gx, G3H, wave, 16, lane);
    tile_gemm<HDIM>(hb[3], wp + OW_G4HH, a.bhh[3], G3H, gh, G3H, wave, 16, lane);
    __syncthreads();
    gru_combine(gx, gh, hb[3], tid, a.outb, t, b0);
    __syncthreads();
  }
}

// ---------------------------------------------------------------------------
// Classification heads kernel: one 16-row tile per WG (8192 WGs x 256 thr).
// ---------------------------------------------------------------------------
struct ClsArgs {
  const __bf16* hin;
  const __bf16* wpool;
  size_t w1off[2], w2off[2], w3off[2], w4off[2];
  const float* b1[2];
  const float* lng[2];
  const float* lnb[2];
  const float* b2[2];
  const float* b3[2];
  const float* b4[2];
  float* out[2];
  int nout[2];
};

__global__ void __launch_bounds__(256) cls_kernel(ClsArgs a) {
  __shared__ __attribute__((aligned(16))) __bf16 inb[16 * HDIM];
  __shared__ __attribute__((aligned(16))) float  fbuf[16 * HDIM];
  __shared__ __attribute__((aligned(16))) __bf16 tb[16 * HDIM];
  const int tid = threadIdx.x, lane = tid & 31, wave = tid >> 5;
  const size_t r0 = (size_t)blockIdx.x * 16;

  for (int i = tid; i < 16 * HDIM; i += 256) inb[i] = a.hin[r0 * HDIM + i];
  __syncthreads();

  for (int hd = 0; hd < 2; ++hd) {
    tile_gemm<HDIM>(inb, a.wpool + a.w1off[hd], a.b1[hd], HDIM, fbuf, HDIM, wave, 8, lane);
    __syncthreads();
    ln_to_bf(fbuf, a.lng[hd], a.lnb[hd], true, tb, wave, 8, lane);
    __syncthreads();
    tile_gemm<HDIM>(tb, a.wpool + a.w2off[hd], a.b2[hd], HDIM, fbuf, HDIM, wave, 8, lane);
    __syncthreads();
    relu_to_bf(fbuf, tb, 16 * HDIM, tid, 256);
    __syncthreads();
    tile_gemm<HDIM>(tb, a.wpool + a.w3off[hd], a.b3[hd], HDIM, fbuf, HDIM, wave, 8, lane);
    __syncthreads();
    relu_to_bf(fbuf, tb, 16 * HDIM, tid, 256);
    __syncthreads();
    tile_gemm<HDIM>(tb, a.wpool + a.w4off[hd], a.b4[hd], a.nout[hd], fbuf, 16, wave, 8, lane);
    __syncthreads();
    const int no = a.nout[hd];
    float* op = a.out[hd];
    for (int i = tid; i < 16 * no; i += 256) {
      const int m = i / no, n = i - m * no;
      op[(r0 + m) * no + n] = fbuf[m * 16 + n];
    }
    __syncthreads();
  }
}

// ---------------------------------------------------------------------------
// Host launcher
// ---------------------------------------------------------------------------
extern "C" void kernel_launch(void* const* d_in, const int* in_sizes, int n_in,
                              void* d_out, int out_size, void* d_ws, size_t ws_size,
                              hipStream_t stream) {
  (void)in_sizes; (void)n_in; (void)out_size; (void)ws_size;

  const float* X = (const float*)d_in[0];
  float* bnscale = (float*)d_ws;
  float* bnshift = bnscale + 64;
  __bf16* pool = (__bf16*)((char*)d_ws + 512);
  __bf16* outb = pool + POOL_ELEMS;

  // ---- weight f32 -> bf16 conversion (pool offsets are compile-time) ----
  struct J { int idx; size_t off; int n; };
  const J jobs[] = {
      {4, OW_G1IH, (int)SZ_IHF}, {5, OW_G1HH, (int)SZ_HH},
      {8, OW_G2IH, (int)SZ_HH},  {9, OW_G2HH, (int)SZ_HH},
      {12, OW_G3IH, (int)SZ_HH}, {13, OW_G3HH, (int)SZ_HH},
      {16, OW_G4IH, (int)SZ_IHF}, {17, OW_G4HH, (int)SZ_HH},
      {26, OW_FC1A, (int)SZ_FC}, {28, OW_FC1B, (int)SZ_FC},
      {30, OW_FC2A, (int)SZ_FC}, {32, OW_FC2B, (int)SZ_FC},
      {34, OW_FC3A, (int)SZ_FC}, {36, OW_FC3B, (int)SZ_FC},
      {38, OW_FC4A, (int)SZ_FC4A}, {40, OW_FC4B, (int)SZ_FC4B},
      {42, OW_C1W1, (int)SZ_FC}, {46, OW_C1W2, (int)SZ_FC}, {48, OW_C1W3, (int)SZ_FC},
      {52, OW_C2W1, (int)SZ_FC}, {56, OW_C2W2, (int)SZ_FC}, {58, OW_C2W3, (int)SZ_FC},
  };
  for (const J& j : jobs) {
    cvt_bf16_kernel<<<(j.n + 255) / 256, 256, 0, stream>>>(
        (const float*)d_in[j.idx], pool + j.off, j.n);
  }
  cvt_pad_kernel<<<16, 256, 0, stream>>>((const float*)d_in[50], pool + OW_C1W4, 7, 256);
  cvt_pad_kernel<<<16, 256, 0, stream>>>((const float*)d_in[60], pool + OW_C2W4, 5, 256);

  // ---- BatchNorm training-mode stats over (S, B) per feature ----
  bn_stats_kernel<<<64, 256, 0, stream>>>(X, (const float*)d_in[2],
                                          (const float*)d_in[3], bnscale, bnshift);

  // ---- persistent scan ----
  ScanArgs sa{};
  sa.X = X; sa.bnscale = bnscale; sa.bnshift = bnshift;
  sa.wpool = pool; sa.outb = outb;
  sa.bih[0] = (const float*)d_in[6];  sa.bhh[0] = (const float*)d_in[7];
  sa.bih[1] = (const float*)d_in[10]; sa.bhh[1] = (const float*)d_in[11];
  sa.bih[2] = (const float*)d_in[14]; sa.bhh[2] = (const float*)d_in[15];
  sa.bih[3] = (const float*)d_in[18]; sa.bhh[3] = (const float*)d_in[19];
  sa.fcb1[0] = (const float*)d_in[27]; sa.fcb2[0] = (const float*)d_in[29];
  sa.fcb1[1] = (const float*)d_in[31]; sa.fcb2[1] = (const float*)d_in[33];
  sa.fcb1[2] = (const float*)d_in[35]; sa.fcb2[2] = (const float*)d_in[37];
  sa.fcb1[3] = (const float*)d_in[39]; sa.fcb2[3] = (const float*)d_in[41];
  sa.lng[0] = (const float*)d_in[20]; sa.lnb[0] = (const float*)d_in[21];
  sa.lng[1] = (const float*)d_in[22]; sa.lnb[1] = (const float*)d_in[23];
  sa.lng[2] = (const float*)d_in[24]; sa.lnb[2] = (const float*)d_in[25];
  scan_kernel<<<BATCH / 16, 512, 0, stream>>>(sa);

  // ---- classification heads ----
  ClsArgs ca{};
  ca.hin = outb; ca.wpool = pool;
  ca.w1off[0] = OW_C1W1; ca.w2off[0] = OW_C1W2; ca.w3off[0] = OW_C1W3; ca.w4off[0] = OW_C1W4;
  ca.w1off[1] = OW_C2W1; ca.w2off[1] = OW_C2W2; ca.w3off[1] = OW_C2W3; ca.w4off[1] = OW_C2W4;
  ca.b1[0] = (const float*)d_in[43]; ca.lng[0] = (const float*)d_in[44];
  ca.lnb[0] = (const float*)d_in[45]; ca.b2[0] = (const float*)d_in[47];
  ca.b3[0] = (const float*)d_in[49]; ca.b4[0] = (const float*)d_in[51];
  ca.b1[1] = (const float*)d_in[53]; ca.lng[1] = (const float*)d_in[54];
  ca.lnb[1] = (const float*)d_in[55]; ca.b2[1] = (const float*)d_in[57];
  ca.b3[1] = (const float*)d_in[59]; ca.b4[1] = (const float*)d_in[61];
  ca.out[0] = (float*)d_out;
  ca.out[1] = (float*)d_out + (size_t)S_LEN * BATCH * 7;
  ca.nout[0] = 7; ca.nout[1] = 5;
  cls_kernel<<<(S_LEN * BATCH) / 16, 256, 0, stream>>>(ca);
}
